// FlatGNN_73048803770590
// MI455X (gfx1250) — matrix operand order repaired and checked by reading, not compiled, not run on hardware
//
#include <hip/hip_runtime.h>

#define NN 16
#define DD 128
#define LLAYERS 3
#define EE 120          // N*(N-1)/2
#define XCOLS (NN + EE) // 136
#define FRAG128 16384   // halves per packed 128x128 B matrix (4 ksteps * 8 ntiles * 32 lanes * 16)
#define SMEM_BYTES 79760

typedef __attribute__((ext_vector_type(16))) _Float16 v16h;
typedef __attribute__((ext_vector_type(8)))  _Float16 v8h;
typedef __attribute__((ext_vector_type(8)))  float    v8f;

__device__ __forceinline__ float siluf(float x) { return x / (1.0f + __expf(-x)); }

__device__ __forceinline__ float wave_sum32(float v) {
#pragma unroll
  for (int m = 16; m >= 1; m >>= 1) v += __shfl_xor(v, m, 32);
  return v;
}

// A-fragment (16x32 f16 tile) from row-major f16 in LDS.
__device__ __forceinline__ v16h load_a_frag(const _Float16* s, int stride, int m0, int k0, int lane) {
  int m = m0 + (lane & 15);
  int hh = lane >> 4;
  v8h lo = *(const v8h*)(s + m * stride + k0 + hh * 16);
  v8h hi = *(const v8h*)(s + m * stride + k0 + hh * 16 + 8);
  v16h r;
#pragma unroll
  for (int i = 0; i < 8; i++) { r[i] = lo[i]; r[i + 8] = hi[i]; }
  return r;
}

// B-fragment pre-packed in global memory (fragment-major).
__device__ __forceinline__ v16h load_b_frag(const _Float16* base, int kstep, int ntile, int ntiles, int lane) {
  return *(const v16h*)(base + ((size_t)(kstep * ntiles + ntile) * 32 + lane) * 16);
}

// C fragment (16x16 f32) -> row-major f32 LDS. lane L: col n0+(L&15), rows r + 8*(L>>4).
__device__ __forceinline__ void store_c_lds(float* dst, int stride, int m0, int n0, v8f c, int lane) {
  int n = n0 + (lane & 15);
  int hh = lane >> 4;
#pragma unroll
  for (int r = 0; r < 8; r++) dst[(m0 + hh * 8 + r) * stride + n] = c[r];
}

__device__ __forceinline__ v8f wmma_f16(v16h a, v16h b, v8f c) {
  return __builtin_amdgcn_wmma_f32_16x16x32_f16(false, a, false, b, (short)0, c, false, false);
}

// ---------------------------------------------------------------------------
// Pack a KxNc f32 row-major matrix into CDNA5 B-fragment layout.
// grid.x = ksteps*ntiles, block = 32 (one wave per fragment).
__global__ void pack_b_frags(const float* __restrict__ src, _Float16* __restrict__ dst,
                             int stride, int ntiles) {
  int blk = blockIdx.x;
  int lane = threadIdx.x;
  int ntile = blk % ntiles;
  int kbase = (blk / ntiles) * 32 + (lane >> 4) * 16;
  int n = ntile * 16 + (lane & 15);
  _Float16* d = dst + ((size_t)blk * 32 + lane) * 16;
#pragma unroll
  for (int e = 0; e < 16; e++) d[e] = (_Float16)src[(size_t)(kbase + e) * stride + n];
}

// ---------------------------------------------------------------------------
// One workgroup (256 threads = 8 wave32) per graph.
__global__ __launch_bounds__(256) void gnn_kernel(
    const float* __restrict__ X,
    const float* __restrict__ encW, const float* __restrict__ encB,
    const float* __restrict__ msgW,  // for w3 rows
    const float* __restrict__ msgB, const float* __restrict__ msgG, const float* __restrict__ msgBe,
    const float* __restrict__ updB, const float* __restrict__ updG, const float* __restrict__ updBe,
    const float* __restrict__ decW1, // for w1c row
    const float* __restrict__ decB1, const float* __restrict__ decB2,
    const float* __restrict__ decW3, const float* __restrict__ decB3,
    const _Float16* __restrict__ frags,
    float* __restrict__ out) {
  extern __shared__ char smem[];
  float* h32   = (float*)smem;            // 16*128
  float* agg32 = h32 + NN * DD;           // 16*128
  float* bufA  = agg32 + NN * DD;         // 16*128  (P / U / R)
  float* bufB  = bufA + NN * DD;          // 16*128  (Q / S)
  float* vec   = bufB + NN * DD;          // 7*128  [w3, mb, mg, mbeta, ub, ug, ubeta]
  float* adj   = vec + 7 * DD;            // 16*16
  float* statP = adj + NN * NN;           // 48: [sum, sumsq, dot_w3] x 16 rows
  float* statQ = statP + 48;              // 48
  float* PQ    = statQ + 48;              // 256: Gram matrix P.Q^T
  float* w3st  = PQ + 256;                // 4: [sum(w3), sum(w3^2), pad, pad]
  _Float16* h16   = (_Float16*)(w3st + 4);      // 16*128
  _Float16* agg16 = h16 + NN * DD;              // 16*128
  _Float16* x1h   = agg16 + NN * DD;            // 128*128 (decoder; p16/q16 scratch in layers)
  _Float16* p16 = x1h;
  _Float16* q16 = x1h + NN * DD;

  const int tid = threadIdx.x;
  const int lane = tid & 31;
  const int wave = tid >> 5;
  const int g = blockIdx.x;
  const float* xrow = X + (size_t)g * XCOLS;

  // ---- encode h, build adj ----
  for (int i = tid; i < NN * DD; i += 256) {
    int m = i >> 7, d = i & 127;
    float v = xrow[m] * encW[d] + encB[d];
    h32[i] = v;
    h16[i] = (_Float16)v;
  }
  {
    int r = tid >> 4, c = tid & 15; // tid in [0,256)
    float v = 0.0f;
    if (r != c) {
      int i0 = r < c ? r : c, j0 = r < c ? c : r;
      int e = i0 * 15 - (i0 * (i0 - 1)) / 2 + (j0 - i0 - 1);
      v = xrow[NN + e];
    }
    adj[tid] = v;
  }
  __syncthreads();

  // ================= message-passing layers =================
  for (int l = 0; l < LLAYERS; l++) {
    // stage per-layer vectors into LDS
    const float* w3row = msgW + (size_t)l * 257 * DD + 256 * DD;
    for (int i = tid; i < 7 * DD; i += 256) {
      int which = i >> 7, d = i & 127;
      float v;
      switch (which) {
        case 0: v = w3row[d]; break;
        case 1: v = msgB[l * DD + d]; break;
        case 2: v = msgG[l * DD + d]; break;
        case 3: v = msgBe[l * DD + d]; break;
        case 4: v = updB[l * DD + d]; break;
        case 5: v = updG[l * DD + d]; break;
        default: v = updBe[l * DD + d]; break;
      }
      vec[i] = v;
    }
    __syncthreads();

    // ---- Phase 1: P = h@W1 + msg_b, Q = h@W2 (wave w owns ntile w of both) ----
    {
      const _Float16* fP = frags + (size_t)(2 * l + 0) * FRAG128;
      const _Float16* fQ = frags + (size_t)(2 * l + 1) * FRAG128;
      v8f cp = {}, cq = {};
#pragma unroll
      for (int k = 0; k < 4; k++) {
        v16h a = load_a_frag(h16, DD, 0, k * 32, lane);
        cp = wmma_f16(a, load_b_frag(fP, k, wave, 8, lane), cp);
        cq = wmma_f16(a, load_b_frag(fQ, k, wave, 8, lane), cq);
      }
      float bb = vec[1 * DD + wave * 16 + (lane & 15)];
#pragma unroll
      for (int r = 0; r < 8; r++) cp[r] += bb;
      store_c_lds(bufA, DD, 0, wave * 16, cp, lane);
      store_c_lds(bufB, DD, 0, wave * 16, cq, lane);
    }
    __syncthreads();

    // ---- Phase 2a: f16 copies of P,Q + decomposed LN statistics ----
    for (int i = tid; i < NN * DD; i += 256) {
      p16[i] = (_Float16)bufA[i];
      q16[i] = (_Float16)bufB[i];
    }
    {
      const int c0 = lane * 4;
      float w0 = vec[c0 + 0], w1 = vec[c0 + 1], w2 = vec[c0 + 2], w3v = vec[c0 + 3];
#pragma unroll
      for (int jj = 0; jj < 4; jj++) {
        int job = wave * 4 + jj;                  // 0..15: P rows, 16..31: Q rows
        const float* src = (job < NN) ? (bufA + job * DD) : (bufB + (job - NN) * DD);
        float v0 = src[c0 + 0], v1 = src[c0 + 1], v2 = src[c0 + 2], v3 = src[c0 + 3];
        float s1 = wave_sum32(v0 + v1 + v2 + v3);
        float s2 = wave_sum32(v0 * v0 + v1 * v1 + v2 * v2 + v3 * v3);
        float sw = wave_sum32(v0 * w0 + v1 * w1 + v2 * w2 + v3 * w3v);
        if (lane == 0) {
          float* st = (job < NN) ? statP : statQ;
          int r = (job < NN) ? job : job - NN;
          st[r] = s1; st[16 + r] = s2; st[32 + r] = sw;
        }
      }
      if (wave == 0) {
        float s1 = wave_sum32(w0 + w1 + w2 + w3v);
        float s2 = wave_sum32(w0 * w0 + w1 * w1 + w2 * w2 + w3v * w3v);
        if (lane == 0) { w3st[0] = s1; w3st[1] = s2; }
      }
    }
    __syncthreads();

    // ---- Phase 2b: Gram matrix PQ = P @ Q^T via WMMA (B = Q^T free from layout) ----
    if (wave == 0) {
      v8f c = {};
#pragma unroll
      for (int k = 0; k < 4; k++) {
        v16h a = load_a_frag(p16, DD, 0, k * 32, lane);
        v16h b = *(const v16h*)(q16 + (lane & 15) * DD + k * 32 + (lane >> 4) * 16);
        c = wmma_f16(a, b, c);
      }
      store_c_lds(PQ, 16, 0, 0, c, lane);
    }
    __syncthreads();

    // ---- Phase 2c: m = silu(LN(P[t]+Q[s]+adj*w3)) with closed-form stats; masked sum ----
    {
      const int c0 = lane * 4;
      const float w3s = w3st[0], w3q = w3st[1];
#pragma unroll
      for (int ti = 0; ti < 2; ti++) {
        int t = wave * 2 + ti;
        float p0 = bufA[t * DD + c0 + 0], p1 = bufA[t * DD + c0 + 1];
        float p2 = bufA[t * DD + c0 + 2], p3 = bufA[t * DD + c0 + 3];
        float w0 = vec[c0 + 0], w1 = vec[c0 + 1], w2 = vec[c0 + 2], w3v = vec[c0 + 3];
        float g0 = vec[2 * DD + c0 + 0], g1 = vec[2 * DD + c0 + 1], g2 = vec[2 * DD + c0 + 2], g3 = vec[2 * DD + c0 + 3];
        float e0 = vec[3 * DD + c0 + 0], e1 = vec[3 * DD + c0 + 1], e2 = vec[3 * DD + c0 + 2], e3 = vec[3 * DD + c0 + 3];
        float sumPt = statP[t], SP2t = statP[16 + t], PWt = statP[32 + t];
        float a0 = 0.f, a1 = 0.f, a2 = 0.f, a3 = 0.f;
        for (int s = 0; s < NN; s++) {
          float aj = adj[t * NN + s];
          float mean = (sumPt + statQ[s] + aj * w3s) * (1.0f / 128.0f);
          float ex2 = (SP2t + statQ[16 + s] + aj * aj * w3q +
                       2.0f * (PQ[t * 16 + s] + aj * PWt + aj * statQ[32 + s])) * (1.0f / 128.0f);
          float var = fmaxf(ex2 - mean * mean, 0.0f);
          float inv = __frsqrt_rn(var + 1e-5f);
          float v0 = p0 + bufB[s * DD + c0 + 0] + aj * w0;
          float v1 = p1 + bufB[s * DD + c0 + 1] + aj * w1;
          float v2 = p2 + bufB[s * DD + c0 + 2] + aj * w2;
          float v3 = p3 + bufB[s * DD + c0 + 3] + aj * w3v;
          float m0 = siluf((v0 - mean) * inv * g0 + e0);
          float m1 = siluf((v1 - mean) * inv * g1 + e1);
          float m2 = siluf((v2 - mean) * inv * g2 + e2);
          float m3 = siluf((v3 - mean) * inv * g3 + e3);
          if (s != t) { a0 += m0; a1 += m1; a2 += m2; a3 += m3; }
        }
        agg32[t * DD + c0 + 0] = a0;
        agg32[t * DD + c0 + 1] = a1;
        agg32[t * DD + c0 + 2] = a2;
        agg32[t * DD + c0 + 3] = a3;
      }
    }
    __syncthreads();
    for (int i = tid; i < NN * DD; i += 256) agg16[i] = (_Float16)agg32[i];
    __syncthreads();

    // ---- Phase 3: U = h@Wu1 + agg@Wu2 + upd_b; h += silu(LN(U)) ----
    {
      const _Float16* f1 = frags + (size_t)(6 + 2 * l) * FRAG128;
      const _Float16* f2 = frags + (size_t)(7 + 2 * l) * FRAG128;
      v8f c = {};
#pragma unroll
      for (int k = 0; k < 4; k++)
        c = wmma_f16(load_a_frag(h16, DD, 0, k * 32, lane), load_b_frag(f1, k, wave, 8, lane), c);
#pragma unroll
      for (int k = 0; k < 4; k++)
        c = wmma_f16(load_a_frag(agg16, DD, 0, k * 32, lane), load_b_frag(f2, k, wave, 8, lane), c);
      float bb = vec[4 * DD + wave * 16 + (lane & 15)];
#pragma unroll
      for (int r = 0; r < 8; r++) c[r] += bb;
      store_c_lds(bufA, DD, 0, wave * 16, c, lane);
    }
    __syncthreads();
    {
      const int c0 = lane * 4;
#pragma unroll
      for (int ri = 0; ri < 2; ri++) {
        int m = wave * 2 + ri;
        float v0 = bufA[m * DD + c0 + 0], v1 = bufA[m * DD + c0 + 1];
        float v2 = bufA[m * DD + c0 + 2], v3 = bufA[m * DD + c0 + 3];
        float mean = wave_sum32(v0 + v1 + v2 + v3) * (1.0f / 128.0f);
        float d0 = v0 - mean, d1 = v1 - mean, d2 = v2 - mean, d3 = v3 - mean;
        float var = wave_sum32(d0 * d0 + d1 * d1 + d2 * d2 + d3 * d3) * (1.0f / 128.0f);
        float inv = __frsqrt_rn(var + 1e-5f);
#pragma unroll
        for (int k = 0; k < 4; k++) {
          float d = (k == 0 ? d0 : k == 1 ? d1 : k == 2 ? d2 : d3);
          float u = siluf(d * inv * vec[5 * DD + c0 + k] + vec[6 * DD + c0 + k]);
          float hn = h32[m * DD + c0 + k] + u;
          h32[m * DD + c0 + k] = hn;
          h16[m * DD + c0 + k] = (_Float16)hn;
        }
      }
    }
    __syncthreads();
  }

  // ================= decoder =================
  // stage w1c = dec_W1[256,:]
  for (int i = tid; i < DD; i += 256) vec[i] = decW1[256 * DD + i];

  // ---- Phase 4: R = h@W1a + b1, S = h@W1b ----
  {
    const _Float16* fR = frags + (size_t)12 * FRAG128;
    const _Float16* fS = frags + (size_t)13 * FRAG128;
    v8f cr = {}, cs = {};
#pragma unroll
    for (int k = 0; k < 4; k++) {
      v16h a = load_a_frag(h16, DD, 0, k * 32, lane);
      cr = wmma_f16(a, load_b_frag(fR, k, wave, 8, lane), cr);
      cs = wmma_f16(a, load_b_frag(fS, k, wave, 8, lane), cs);
    }
    float bb = decB1[wave * 16 + (lane & 15)];
#pragma unroll
    for (int r = 0; r < 8; r++) cr[r] += bb;
    store_c_lds(bufA, DD, 0, wave * 16, cr, lane);
    store_c_lds(bufB, DD, 0, wave * 16, cs, lane);
  }
  __syncthreads();

  // ---- Phase 5: x1[e] = silu(R[i] + S[j] + J*w1c), 15 edges per wave; pad rows 120..127 ----
  for (int i = tid; i < 8 * DD; i += 256) x1h[EE * DD + i] = (_Float16)0.0f;
  {
    const int c0 = lane * 4;
    for (int ei = 0; ei < 15; ei++) {
      int e = wave * 15 + ei;
      int i0 = 0, rem = e;
      while (rem >= 15 - i0) { rem -= 15 - i0; i0++; }
      int j0 = i0 + 1 + rem;
      float Jv = adj[i0 * NN + j0];
#pragma unroll
      for (int k = 0; k < 4; k++) {
        float v = bufA[i0 * DD + c0 + k] + bufB[j0 * DD + c0 + k] + Jv * vec[c0 + k];
        x1h[e * DD + c0 + k] = (_Float16)siluf(v);
      }
    }
  }
  __syncthreads();

  // ---- Phase 6: x2 = silu(x1@W2 + b2); out = tanh(x2@W3 + b3). wave w owns mtile w. ----
  {
    const _Float16* fW2 = frags + (size_t)14 * FRAG128;
    const int m0 = wave * 16;
    v8f acc[4];
#pragma unroll
    for (int nt = 0; nt < 4; nt++) {
      v8f c = {};
#pragma unroll
      for (int k = 0; k < 4; k++)
        c = wmma_f16(load_a_frag(x1h, DD, m0, k * 32, lane), load_b_frag(fW2, k, nt, 4, lane), c);
      float b2 = decB2[nt * 16 + (lane & 15)];
#pragma unroll
      for (int r = 0; r < 8; r++) c[r] = siluf(c[r] + b2);
      acc[nt] = c;
    }
    float w3d[4];
#pragma unroll
    for (int nt = 0; nt < 4; nt++) w3d[nt] = decW3[nt * 16 + (lane & 15)];
    float part[8];
#pragma unroll
    for (int r = 0; r < 8; r++)
      part[r] = acc[0][r] * w3d[0] + acc[1][r] * w3d[1] + acc[2][r] * w3d[2] + acc[3][r] * w3d[3];
#pragma unroll
    for (int mask = 8; mask >= 1; mask >>= 1)
#pragma unroll
      for (int r = 0; r < 8; r++) part[r] += __shfl_xor(part[r], mask, 32);
    float b3 = decB3[0];
    if ((lane & 15) == 0) {
      int hh = lane >> 4;
#pragma unroll
      for (int r = 0; r < 8; r++) {
        int e = m0 + hh * 8 + r;
        if (e < EE) out[(size_t)g * EE + e] = tanhf(part[r] + b3);
      }
    }
  }
}

// ---------------------------------------------------------------------------
extern "C" void kernel_launch(void* const* d_in, const int* in_sizes, int n_in,
                              void* d_out, int out_size, void* d_ws, size_t ws_size,
                              hipStream_t stream) {
  const float* X     = (const float*)d_in[0];
  const float* encW  = (const float*)d_in[1];
  const float* encB  = (const float*)d_in[2];
  const float* msgW  = (const float*)d_in[3];
  const float* msgB  = (const float*)d_in[4];
  const float* msgG  = (const float*)d_in[5];
  const float* msgBe = (const float*)d_in[6];
  const float* updW  = (const float*)d_in[7];
  const float* updB  = (const float*)d_in[8];
  const float* updG  = (const float*)d_in[9];
  const float* updBe = (const float*)d_in[10];
  const float* decW1 = (const float*)d_in[11];
  const float* decB1 = (const float*)d_in[12];
  const float* decW2 = (const float*)d_in[13];
  const float* decB2 = (const float*)d_in[14];
  const float* decW3 = (const float*)d_in[15];
  const float* decB3 = (const float*)d_in[16];
  _Float16* frags = (_Float16*)d_ws;
  float* out = (float*)d_out;
  const int B = in_sizes[0] / XCOLS;

  // Pack all WMMA B-operand weights into fragment layout (all K=128 slices).
  for (int l = 0; l < LLAYERS; l++) {
    pack_b_frags<<<32, 32, 0, stream>>>(msgW + (size_t)l * 257 * DD,            frags + (size_t)(2 * l + 0) * FRAG128, DD, 8);
    pack_b_frags<<<32, 32, 0, stream>>>(msgW + (size_t)l * 257 * DD + 128 * DD, frags + (size_t)(2 * l + 1) * FRAG128, DD, 8);
    pack_b_frags<<<32, 32, 0, stream>>>(updW + (size_t)l * 256 * DD,            frags + (size_t)(6 + 2 * l) * FRAG128, DD, 8);
    pack_b_frags<<<32, 32, 0, stream>>>(updW + (size_t)l * 256 * DD + 128 * DD, frags + (size_t)(7 + 2 * l) * FRAG128, DD, 8);
  }
  pack_b_frags<<<32, 32, 0, stream>>>(decW1,            frags + (size_t)12 * FRAG128, DD, 8);
  pack_b_frags<<<32, 32, 0, stream>>>(decW1 + 128 * DD, frags + (size_t)13 * FRAG128, DD, 8);
  pack_b_frags<<<16, 32, 0, stream>>>(decW2,            frags + (size_t)14 * FRAG128, 64, 4);

  gnn_kernel<<<B, 256, SMEM_BYTES, stream>>>(
      X, encW, encB, msgW, msgB, msgG, msgBe, updB, updG, updBe,
      decW1, decB1, decB2, decW3, decB3, frags, out);
}
